// SEBlock_62732292325378
// MI455X (gfx1250) — matrix operand order repaired and checked by reading, not compiled, not run on hardware
//
#include <hip/hip_runtime.h>
#include <math.h>

// Problem constants from the reference
#define B_  64
#define C_  256
#define HW_ 1024

typedef __attribute__((ext_vector_type(16))) __bf16 v16bf;
typedef __attribute__((ext_vector_type(8)))  float  v8f;
typedef __attribute__((ext_vector_type(4)))  float  nf4;   // native vec for NT ld/st

// ---------------------------------------------------------------------------
// Kernel 1: single pass over x computes
//   ch_sq[b][c]  = mean over hw of x[b][c][hw]   (sSE squeeze)
//   sp_sq[b][hw] = mean over c  of x[b][c][hw]   (cSE squeeze)
// One block per batch, 256 threads. Each thread owns 4 hw columns (register
// accumulation, coalesced float4 reads); per-channel sums via wave32 shuffle
// reduction into LDS partials (8 waves -> 8 partials per channel).
// ---------------------------------------------------------------------------
__global__ void se_reduce(const float* __restrict__ x,
                          float* __restrict__ ch_sq,
                          float* __restrict__ sp_sq) {
  const int b    = blockIdx.x;
  const int t    = threadIdx.x;      // 0..255
  const int lane = t & 31;
  const int wave = t >> 5;           // 0..7
  const float4* xb4 = (const float4*)(x + (size_t)b * C_ * HW_);

  __shared__ float chpart[C_ * 8];

  float4 sp = make_float4(0.f, 0.f, 0.f, 0.f);
  for (int c = 0; c < C_; ++c) {
    float4 v = xb4[c * (HW_ / 4) + t];
    sp.x += v.x; sp.y += v.y; sp.z += v.z; sp.w += v.w;
    float s = v.x + v.y + v.z + v.w;
#pragma unroll
    for (int off = 16; off > 0; off >>= 1) s += __shfl_xor(s, off, 32);
    if (lane == 0) chpart[c * 8 + wave] = s;   // partial over 128 hw
  }
  __syncthreads();

  // thread t finalizes channel t
  float cs = 0.f;
#pragma unroll
  for (int j = 0; j < 8; ++j) cs += chpart[t * 8 + j];
  ch_sq[b * C_ + t] = cs * (1.f / (float)HW_);

  float4 o;
  o.x = sp.x * (1.f / (float)C_);
  o.y = sp.y * (1.f / (float)C_);
  o.z = sp.z * (1.f / (float)C_);
  o.w = sp.w * (1.f / (float)C_);
  ((float4*)sp_sq)[b * (HW_ / 4) + t] = o;
}

// ---------------------------------------------------------------------------
// Kernel 2: generic  out = act(A @ W^T + bias)  via v_wmma_f32_16x16x32_bf16.
//   A   : M x K  (f32, row-major)
//   W   : N x K  (f32, row-major; torch Linear weight) => B[k][n] = W[n][k]
//   out : M x N  (f32)
// One 16x16 output tile per wave, 4 waves per block along N.
// ACT: 0 = ReLU, 1 = sigmoid.
// ---------------------------------------------------------------------------
template <int ACT>
__global__ void gemm_bt_wmma(const float* __restrict__ A,
                             const float* __restrict__ W,
                             const float* __restrict__ bias,
                             float* __restrict__ out,
                             int M, int N, int K) {
  const int wave = threadIdx.x >> 5;          // 0..3
  const int lane = threadIdx.x & 31;
  const int half = lane >> 4;                 // 0 or 1
  const int idx  = lane & 15;                 // A: row-in-tile, B: col-in-tile
  const int n0   = (blockIdx.x * 4 + wave) * 16;
  const int m0   = blockIdx.y * 16;

  const float* Arow = A + (size_t)(m0 + idx) * K;  // lane's A row (M = m0+idx)
  const float* Wrow = W + (size_t)(n0 + idx) * K;  // lane's B col (N = n0+idx)

  v8f acc = {};
  for (int k = 0; k < K; k += 32) {
    // ---- A fragment: 16x32 bf16. VGPRs 0-7: K = half*16 + 0..15 (contiguous,
    //                              matching B).
    float af[16];
    *(float4*)(af + 0)  = *(const float4*)(Arow + k + half * 16);
    *(float4*)(af + 4)  = *(const float4*)(Arow + k + half * 16 + 4);
    *(float4*)(af + 8)  = *(const float4*)(Arow + k + half * 16 + 8);
    *(float4*)(af + 12) = *(const float4*)(Arow + k + half * 16 + 12);
    // ---- B fragment: 32x16 bf16. VGPRs 0-7: K = half*16 + 0..15 (contiguous
    //      in W's row since B[k][n] = W[n][k]).
    float wf[16];
    *(float4*)(wf + 0)  = *(const float4*)(Wrow + k + half * 16);
    *(float4*)(wf + 4)  = *(const float4*)(Wrow + k + half * 16 + 4);
    *(float4*)(wf + 8)  = *(const float4*)(Wrow + k + half * 16 + 8);
    *(float4*)(wf + 12) = *(const float4*)(Wrow + k + half * 16 + 12);

    v16bf a, bm;
#pragma unroll
    for (int j = 0; j < 16; ++j) {
      a[j]  = (__bf16)af[j];
      bm[j] = (__bf16)wf[j];
    }
    // (neg_a, A, neg_b, B, c_mod, C, reuse_a, reuse_b)
    acc = __builtin_amdgcn_wmma_f32_16x16x32_bf16(false, a, false, bm,
                                                  (short)0, acc, false, false);
  }

  // D layout: VGPR r -> row m0 + half*8 + r, column n0 + idx.
  const int   ncol = n0 + idx;
  const float bv   = bias[ncol];
#pragma unroll
  for (int r = 0; r < 8; ++r) {
    const int mrow = m0 + half * 8 + r;
    float v = acc[r] + bv;
    if (ACT == 0) v = v > 0.f ? v : 0.f;
    else          v = 1.f / (1.f + __expf(-v));
    out[(size_t)mrow * N + ncol] = v;
  }
}

// ---------------------------------------------------------------------------
// Kernel 3: fused gating  out = max(target * g_c[b,c], target * g_s[b,hw])
// float4 per thread; non-temporal load of target + non-temporal store of out
// (201 MB streaming working set slightly exceeds the 192 MB L2).
// ---------------------------------------------------------------------------
__global__ void se_combine(const float* __restrict__ target,
                           const float* __restrict__ g_c,
                           const float* __restrict__ g_s,
                           float* __restrict__ out) {
  const unsigned i   = blockIdx.x * blockDim.x + threadIdx.x; // over B*C*HW/4
  const unsigned hw4 = i & 255;          // HW/4 = 256
  const unsigned c   = (i >> 8) & 255;
  const unsigned b   = i >> 16;

  nf4   t  = __builtin_nontemporal_load((const nf4*)target + i);
  float gc = g_c[b * C_ + c];
  float4 gs = ((const float4*)g_s)[b * (HW_ / 4) + hw4];

  nf4 o;
  o.x = fmaxf(t.x * gc, t.x * gs.x);
  o.y = fmaxf(t.y * gc, t.y * gs.y);
  o.z = fmaxf(t.z * gc, t.z * gs.z);
  o.w = fmaxf(t.w * gc, t.w * gs.w);
  __builtin_nontemporal_store(o, (nf4*)out + i);
}

// ---------------------------------------------------------------------------
extern "C" void kernel_launch(void* const* d_in, const int* in_sizes, int n_in,
                              void* d_out, int out_size, void* d_ws, size_t ws_size,
                              hipStream_t stream) {
  const float* x      = (const float*)d_in[0];
  const float* target = (const float*)d_in[1];
  const float* sW1    = (const float*)d_in[2];  // (1024, 256)
  const float* sb1    = (const float*)d_in[3];
  const float* sW2    = (const float*)d_in[4];  // (256, 1024)
  const float* sb2    = (const float*)d_in[5];
  const float* cW1    = (const float*)d_in[6];  // (4096, 1024)
  const float* cb1    = (const float*)d_in[7];
  const float* cW2    = (const float*)d_in[8];  // (1024, 4096)
  const float* cb2    = (const float*)d_in[9];
  float* out = (float*)d_out;

  // Scratch layout in d_ws (~1.9 MB total)
  float* ws    = (float*)d_ws;
  float* ch_sq = ws;                    // 64*256
  float* sp_sq = ch_sq + B_ * C_;       // 64*1024
  float* h_s   = sp_sq + B_ * HW_;      // 64*1024   relu(ch_sq @ sW1^T + sb1)
  float* g_c   = h_s   + B_ * 4 * C_;   // 64*256    sigmoid(h_s @ sW2^T + sb2)
  float* h_c   = g_c   + B_ * C_;       // 64*4096   relu(sp_sq @ cW1^T + cb1)
  float* g_s   = h_c   + B_ * 4 * HW_;  // 64*1024   sigmoid(h_c @ cW2^T + cb2)

  // 1) both squeezes in one pass over x
  se_reduce<<<B_, 256, 0, stream>>>(x, ch_sq, sp_sq);

  const dim3 blk(128);
  // 2) sSE MLP:  (64x256)@(256x1024) -> relu -> (64x1024)@(1024x256) -> sigmoid
  gemm_bt_wmma<0><<<dim3((4 * C_) / 64, B_ / 16), blk, 0, stream>>>(
      ch_sq, sW1, sb1, h_s, B_, 4 * C_, C_);
  gemm_bt_wmma<1><<<dim3(C_ / 64, B_ / 16), blk, 0, stream>>>(
      h_s, sW2, sb2, g_c, B_, C_, 4 * C_);
  // 3) cSE MLP:  (64x1024)@(1024x4096) -> relu -> (64x4096)@(4096x1024) -> sigmoid
  gemm_bt_wmma<0><<<dim3((4 * HW_) / 64, B_ / 16), blk, 0, stream>>>(
      sp_sq, cW1, cb1, h_c, B_, 4 * HW_, HW_);
  gemm_bt_wmma<1><<<dim3(HW_ / 64, B_ / 16), blk, 0, stream>>>(
      h_c, cW2, cb2, g_s, B_, HW_, 4 * HW_);

  // 4) fused gating + elementwise max
  const int n4 = B_ * C_ * (HW_ / 4);   // 4,194,304 float4s
  se_combine<<<n4 / 256, 256, 0, stream>>>(target, g_c, g_s, out);
}
